// ChitDecoder_10909216932093
// MI455X (gfx1250) — matrix operand order repaired and checked by reading, not compiled, not run on hardware
//
#include <hip/hip_runtime.h>

// ---------------------------------------------------------------------------
// ChitDecoder: LSTM(300) over [B=32,T=128,D=300] + TimeDistributed Dense(32000)
// CDNA5 / gfx1250, wave32, bf16 WMMA (v_wmma_f32_16x16x32_bf16) everywhere.
// ---------------------------------------------------------------------------

typedef __attribute__((ext_vector_type(16))) __bf16 v16bf;
typedef __attribute__((ext_vector_type(8)))  float  v8f;

#define B_    32
#define T_    128
#define D_    300
#define U_    300
#define G4_   1200          // 4*U
#define V_    32000
#define KT_   10            // K padded to 320 -> 10 k-tiles of 32 (bf16 WMMA)
#define MT_X  256           // (B*T)/16 M-tiles
#define NT_WK 75            // 1200/16 N-tiles
#define NT_WR 76            // 19 u-tiles * 4 gates (gate-packed Wr)
#define NT_WD 2000          // 32000/16 N-tiles
#define UT_   19            // ceil(300/16)

static __device__ __forceinline__ float sigmoidf_(float x) {
    return 1.0f / (1.0f + __expf(-x));
}

// A-fragment scatter: 16-bit A matrix 16x32 layout (ISA 7.12.2):
// lane = (M%16) + 16*lanegroup ; per-lane halves: k_local = 2*(v%4)+16*(v/4)+8*lg+half
static __device__ __forceinline__ void store_a_frag(__bf16* buf, int mt, int mrow,
                                                    int k, __bf16 val) {
    int kt   = k >> 5;
    int kl   = k & 31;
    int lg   = (kl >> 3) & 1;
    int base = kl - (lg << 3);               // {0..7} U {16..23}
    int v    = ((base & 15) >> 1) + ((base >> 4) << 2);
    int half = base & 1;
    int lane = mrow + (lg << 4);
    buf[((((long)mt * KT_ + kt) * 32 + lane) << 4) + (v << 1) + half] = val;
}

// ---------------------------------------------------------------------------
// Packing kernels: fp32 row-major -> bf16 WMMA fragment buffers
// ---------------------------------------------------------------------------

// A-matrix pack: X is [Mtiles*16, Kreal] row-major; K padded with zeros to KT_*32.
__global__ void pack_a_kernel(const float* __restrict__ X, __bf16* __restrict__ out,
                              int Kreal, long total) {
    for (long i = (long)blockIdx.x * blockDim.x + threadIdx.x; i < total;
         i += (long)gridDim.x * blockDim.x) {
        int h    = (int)(i & 15);
        int lane = (int)((i >> 4) & 31);
        int kt   = (int)((i >> 9) % KT_);
        int mt   = (int)(i / ((long)KT_ * 512));
        int m    = mt * 16 + (lane & 15);
        int lg   = lane >> 4;
        int v    = h >> 1, half = h & 1;
        int kl   = ((v & 3) << 1) + ((v >> 2) << 4) + (lg << 3) + half;
        int k    = kt * 32 + kl;
        float val = (k < Kreal) ? X[(long)m * Kreal + k] : 0.0f;
        out[i] = (__bf16)val;
    }
}

// B-matrix pack: W is [Kreal, N] row-major. B layout: lane%16 = column,
// per-lane halves are K = kt*32 + 16*(h/8) + 8*(lane/16) + h%8.
__global__ void pack_b_kernel(const float* __restrict__ W, __bf16* __restrict__ out,
                              int Kreal, int N, long total) {
    for (long i = (long)blockIdx.x * blockDim.x + threadIdx.x; i < total;
         i += (long)gridDim.x * blockDim.x) {
        int h    = (int)(i & 15);
        int lane = (int)((i >> 4) & 31);
        int kt   = (int)((i >> 9) % KT_);
        int nt   = (int)(i / ((long)KT_ * 512));
        int n    = nt * 16 + (lane & 15);
        int k    = kt * 32 + ((h >> 3) << 4) + ((lane >> 4) << 3) + (h & 7);
        float val = (k < Kreal) ? W[(long)k * N + n] : 0.0f;
        out[i] = (__bf16)val;
    }
}

// Wr pack with gate-interleaved N-tiles: ntile' = ut*4 + gate, covering
// original columns gate*300 + ut*16 + col (zero-padded past u=300 / k=300).
__global__ void pack_wr_kernel(const float* __restrict__ Wr, __bf16* __restrict__ out,
                               long total) {
    for (long i = (long)blockIdx.x * blockDim.x + threadIdx.x; i < total;
         i += (long)gridDim.x * blockDim.x) {
        int h    = (int)(i & 15);
        int lane = (int)((i >> 4) & 31);
        int kt   = (int)((i >> 9) % KT_);
        int ntp  = (int)(i / ((long)KT_ * 512));
        int ut   = ntp >> 2;
        int g    = ntp & 3;
        int u    = ut * 16 + (lane & 15);
        int k    = kt * 32 + ((h >> 3) << 4) + ((lane >> 4) << 3) + (h & 7);
        float val = (k < U_ && u < U_) ? Wr[(long)k * G4_ + g * U_ + u] : 0.0f;
        out[i] = (__bf16)val;
    }
}

__global__ void zero_kernel(float4* __restrict__ p, long n4) {
    for (long i = (long)blockIdx.x * blockDim.x + threadIdx.x; i < n4;
         i += (long)gridDim.x * blockDim.x) {
        p[i] = make_float4(0.f, 0.f, 0.f, 0.f);
    }
}

// ---------------------------------------------------------------------------
// Register-blocked, software-pipelined WMMA GEMM on pre-packed fragments:
// out[M,N] = A*B (+ bias). Each wave computes NTPW consecutive 16x16 N-tiles
// of one M-tile (A frag reused NTPW x per k-step), and prefetches the next
// k-step's fragments before issuing the current WMMAs so load latency hides
// under matrix issue. No divergent guards -> EXEC all-ones around every WMMA.
// ---------------------------------------------------------------------------
template <int NTPW>
__global__ void gemm_wmma_kernel(const __bf16* __restrict__ Af,
                                 const __bf16* __restrict__ Bf,
                                 const float* __restrict__ bias,
                                 float* __restrict__ out,
                                 int ldN) {
    int lane  = threadIdx.x & 31;
    int w     = threadIdx.x >> 5;
    int waves = blockDim.x >> 5;
    int nt0   = (blockIdx.x * waves + w) * NTPW;
    int mt    = blockIdx.y;

    v8f acc[NTPW];
#pragma unroll
    for (int j = 0; j < NTPW; ++j)
        acc[j] = (v8f){0.f, 0.f, 0.f, 0.f, 0.f, 0.f, 0.f, 0.f};

    const v16bf* ap = (const v16bf*)(Af + (((long)mt * KT_) * 32 + lane) * 16);
    const v16bf* bp[NTPW];
#pragma unroll
    for (int j = 0; j < NTPW; ++j)
        bp[j] = (const v16bf*)(Bf + (((long)(nt0 + j) * KT_) * 32 + lane) * 16);

    // prologue: fetch k-step 0
    v16bf a = ap[0];
    v16bf b[NTPW];
#pragma unroll
    for (int j = 0; j < NTPW; ++j) b[j] = bp[j][0];

#pragma unroll
    for (int kt = 0; kt < KT_; ++kt) {
        v16bf a_n = a;
        v16bf b_n[NTPW];
        if (kt + 1 < KT_) {                       // compile-time after unroll
            a_n = ap[(kt + 1) * 32];
#pragma unroll
            for (int j = 0; j < NTPW; ++j) b_n[j] = bp[j][(kt + 1) * 32];
        } else {
#pragma unroll
            for (int j = 0; j < NTPW; ++j) b_n[j] = b[j];
        }
#pragma unroll
        for (int j = 0; j < NTPW; ++j)
            acc[j] = __builtin_amdgcn_wmma_f32_16x16x32_bf16(false, a, false, b[j],
                                                             (short)0, acc[j], false, false);
        a = a_n;
#pragma unroll
        for (int j = 0; j < NTPW; ++j) b[j] = b_n[j];
    }

    // C/D layout: N = lane%16, M = r + 8*(lane/16)
    int rowbase = mt * 16 + ((lane >> 4) << 3);
#pragma unroll
    for (int j = 0; j < NTPW; ++j) {
        int col  = (nt0 + j) * 16 + (lane & 15);
        float bv = bias ? bias[col] : 0.0f;
#pragma unroll
        for (int r = 0; r < 8; ++r)
            out[(long)(rowbase + r) * ldN + col] = acc[j][r] + bv;
    }
}

// ---------------------------------------------------------------------------
// One LSTM time step, fully fused: g = xz_t + h@Wr ; gates ; c,h update ;
// scatter h (bf16) into A-fragment buffers for next step and for dense GEMM.
// grid = 19 u-tiles, block = 64 (wave 0 -> rows 0..15, wave 1 -> rows 16..31).
// A fragment reused by the 4 gate WMMAs; next k-step prefetched (this kernel
// is the serialized latency chain, 128 back-to-back launches).
// ---------------------------------------------------------------------------
__global__ void lstm_step_kernel(const __bf16* __restrict__ hAin,
                                 __bf16* __restrict__ hAout,
                                 const __bf16* __restrict__ WrF,
                                 const float* __restrict__ xz,
                                 float* __restrict__ c,
                                 float* __restrict__ hf,
                                 __bf16* __restrict__ seqA,
                                 int t) {
    int lane = threadIdx.x & 31;
    int w    = threadIdx.x >> 5;   // M-tile (row group)
    int ut   = blockIdx.x;

    v8f acc[4];
#pragma unroll
    for (int g = 0; g < 4; ++g)
        acc[g] = (v8f){0.f, 0.f, 0.f, 0.f, 0.f, 0.f, 0.f, 0.f};

    const v16bf* ap = (const v16bf*)(hAin + (((long)w * KT_) * 32 + lane) * 16);
    const v16bf* bp[4];
#pragma unroll
    for (int g = 0; g < 4; ++g)
        bp[g] = (const v16bf*)(WrF + ((((long)(ut * 4 + g) * KT_) * 32 + lane) * 16));

    v16bf a = ap[0];
    v16bf b[4];
#pragma unroll
    for (int g = 0; g < 4; ++g) b[g] = bp[g][0];

#pragma unroll
    for (int kt = 0; kt < KT_; ++kt) {
        v16bf a_n = a;
        v16bf b_n[4];
        if (kt + 1 < KT_) {
            a_n = ap[(kt + 1) * 32];
#pragma unroll
            for (int g = 0; g < 4; ++g) b_n[g] = bp[g][(kt + 1) * 32];
        } else {
#pragma unroll
            for (int g = 0; g < 4; ++g) b_n[g] = b[g];
        }
#pragma unroll
        for (int g = 0; g < 4; ++g)
            acc[g] = __builtin_amdgcn_wmma_f32_16x16x32_bf16(false, a, false, b[g],
                                                             (short)0, acc[g], false, false);
        a = a_n;
#pragma unroll
        for (int g = 0; g < 4; ++g) b[g] = b_n[g];
    }

    int col = lane & 15;
    int u   = ut * 16 + col;
    if (u < U_) {
        int rb = (lane >> 4) << 3;
#pragma unroll
        for (int r = 0; r < 8; ++r) {
            int  b_ = w * 16 + rb + r;                 // batch row 0..31
            long R  = (long)b_ * T_ + t;               // flat (b,t) row
            const float* z = xz + R * G4_;
            float iv = sigmoidf_(acc[0][r] + z[u]);
            float fv = sigmoidf_(acc[1][r] + z[U_ + u]);
            float gv = tanhf(acc[2][r] + z[2 * U_ + u]);
            float ov = sigmoidf_(acc[3][r] + z[3 * U_ + u]);
            float cn = fv * c[b_ * U_ + u] + iv * gv;
            float hn = ov * tanhf(cn);
            c[b_ * U_ + u]  = cn;
            hf[b_ * U_ + u] = hn;
            __bf16 hb = (__bf16)hn;
            store_a_frag(hAout, b_ >> 4, b_ & 15, u, hb);     // next-step A frags
            int Rr = b_ * T_ + t;
            store_a_frag(seqA, Rr >> 4, Rr & 15, u, hb);      // dense-GEMM A frags
        }
    }
}

// ---------------------------------------------------------------------------
// Softmax: per-row max + sum(exp), then in-place normalize of d_out.
// ---------------------------------------------------------------------------
__global__ void rowstat_kernel(const float* __restrict__ lg, float* __restrict__ stats) {
    __shared__ float red[256];
    int row = blockIdx.x;
    int tid = threadIdx.x;
    const float* p = lg + (long)row * V_;
    float m = -3.4e38f;
    for (int j = tid; j < V_; j += 256) m = fmaxf(m, p[j]);
    red[tid] = m; __syncthreads();
    for (int s = 128; s > 0; s >>= 1) {
        if (tid < s) red[tid] = fmaxf(red[tid], red[tid + s]);
        __syncthreads();
    }
    m = red[0]; __syncthreads();
    float sum = 0.f;
    for (int j = tid; j < V_; j += 256) sum += __expf(p[j] - m);
    red[tid] = sum; __syncthreads();
    for (int s = 128; s > 0; s >>= 1) {
        if (tid < s) red[tid] += red[tid + s];
        __syncthreads();
    }
    if (tid == 0) { stats[row * 2] = m; stats[row * 2 + 1] = red[0]; }
}

__global__ void softmax_norm_kernel(float* __restrict__ out,
                                    const float* __restrict__ stats, long total) {
    for (long i = (long)blockIdx.x * blockDim.x + threadIdx.x; i < total;
         i += (long)gridDim.x * blockDim.x) {
        long row = i / V_;
        float m = stats[row * 2];
        float s = stats[row * 2 + 1];
        out[i] = __expf(out[i] - m) * (1.0f / s);
    }
}

__global__ void copy_h_kernel(const float* __restrict__ hf, float* __restrict__ dst) {
    int i = blockIdx.x * blockDim.x + threadIdx.x;
    if (i < B_ * U_) dst[i] = hf[i];
}

// ---------------------------------------------------------------------------
// Launch
// ---------------------------------------------------------------------------
extern "C" void kernel_launch(void* const* d_in, const int* in_sizes, int n_in,
                              void* d_out, int out_size, void* d_ws, size_t ws_size,
                              hipStream_t stream) {
    const float* x    = (const float*)d_in[0];   // [32,128,300]
    const float* Wk   = (const float*)d_in[1];   // [300,1200]
    const float* Wr   = (const float*)d_in[2];   // [300,1200]
    const float* bias = (const float*)d_in[3];   // [1200]
    const float* Wd   = (const float*)d_in[4];   // [300,32000]
    const float* bd   = (const float*)d_in[5];   // [32000]
    float* out = (float*)d_out;                  // probs [32,128,32000] ++ h_T [32,300]

    char* ws = (char*)d_ws;
    size_t o = 0;
    auto alloc = [&](size_t bytes) { size_t r = o; o = (o + bytes + 255) & ~(size_t)255; return r; };

    // Region that must start zeroed each launch (frag pads + initial h/c state):
    size_t seqA_off = alloc((size_t)MT_X * KT_ * 512 * 2);  // seq A-frags (bf16)
    size_t hA0_off  = alloc((size_t)2 * KT_ * 512 * 2);     // h A-frags ping
    size_t hA1_off  = alloc((size_t)2 * KT_ * 512 * 2);     // h A-frags pong
    size_t c_off    = alloc((size_t)B_ * U_ * 4);           // cell state fp32
    size_t zero_end = o;
    size_t xA_off   = alloc((size_t)MT_X * KT_ * 512 * 2);  // x A-frags
    size_t wkF_off  = alloc((size_t)NT_WK * KT_ * 512 * 2); // Wk B-frags
    size_t wrF_off  = alloc((size_t)NT_WR * KT_ * 512 * 2); // Wr B-frags (gate-packed)
    size_t wdF_off  = alloc((size_t)NT_WD * KT_ * 512 * 2); // Wd B-frags
    size_t xz_off   = alloc((size_t)B_ * T_ * G4_ * 4);     // xz fp32 [4096,1200]
    size_t hf_off   = alloc((size_t)B_ * U_ * 4);           // h fp32 (for h_T)
    size_t st_off   = alloc((size_t)B_ * T_ * 2 * 4);       // softmax stats
    (void)ws_size; (void)in_sizes; (void)n_in; (void)out_size;

    __bf16* seqA = (__bf16*)(ws + seqA_off);
    __bf16* hA0  = (__bf16*)(ws + hA0_off);
    __bf16* hA1  = (__bf16*)(ws + hA1_off);
    float*  cst  = (float*)(ws + c_off);
    __bf16* xA   = (__bf16*)(ws + xA_off);
    __bf16* wkF  = (__bf16*)(ws + wkF_off);
    __bf16* wrF  = (__bf16*)(ws + wrF_off);
    __bf16* wdF  = (__bf16*)(ws + wdF_off);
    float*  xz   = (float*)(ws + xz_off);
    float*  hf   = (float*)(ws + hf_off);
    float*  st   = (float*)(ws + st_off);

    // 0) zero seqA / hA / c
    long n4 = (long)(zero_end / 16);
    zero_kernel<<<dim3((unsigned)((n4 + 255) / 256)), dim3(256), 0, stream>>>((float4*)ws, n4);

    // 1) pack activations + weights into WMMA fragment layouts
    {
        long tot = (long)MT_X * KT_ * 512;
        pack_a_kernel<<<dim3((unsigned)((tot + 255) / 256)), dim3(256), 0, stream>>>(x, xA, D_, tot);
    }
    {
        long tot = (long)NT_WK * KT_ * 512;
        pack_b_kernel<<<dim3((unsigned)((tot + 255) / 256)), dim3(256), 0, stream>>>(Wk, wkF, D_, G4_, tot);
    }
    {
        long tot = (long)NT_WR * KT_ * 512;
        pack_wr_kernel<<<dim3((unsigned)((tot + 255) / 256)), dim3(256), 0, stream>>>(Wr, wrF, tot);
    }
    {
        long tot = (long)NT_WD * KT_ * 512;
        pack_b_kernel<<<dim3((unsigned)((tot + 255) / 256)), dim3(256), 0, stream>>>(Wd, wdF, U_, V_, tot);
    }

    // 2) input projection: xz = x @ Wk + b
    //    75 N-tiles = 25 blocks x 1 wave x 3 N-tiles/wave
    gemm_wmma_kernel<3><<<dim3(25, MT_X), dim3(32), 0, stream>>>(xA, wkF, bias, xz, G4_);

    // 3) LSTM recurrence: 128 fused step kernels, ping-pong h fragments
    for (int t = 0; t < T_; ++t) {
        const __bf16* hin = (t & 1) ? hA1 : hA0;
        __bf16* hout      = (t & 1) ? hA0 : hA1;
        lstm_step_kernel<<<dim3(UT_), dim3(64), 0, stream>>>(hin, hout, wrF, xz, cst, hf, seqA, t);
    }

    // 4) dense: logits = seq @ Wd + bd -> straight into d_out
    //    2000 N-tiles = 125 blocks x 4 waves x 4 N-tiles/wave
    gemm_wmma_kernel<4><<<dim3(125, MT_X), dim3(128), 0, stream>>>(seqA, wdF, bd, out, V_);

    // 5) softmax over each of the 4096 rows, in place on d_out
    rowstat_kernel<<<dim3(B_ * T_), dim3(256), 0, stream>>>(out, st);
    {
        long tot = (long)B_ * T_ * V_;
        softmax_norm_kernel<<<dim3(131072), dim3(256), 0, stream>>>(out, st, tot);
    }

    // 6) h_T tail output
    copy_h_kernel<<<dim3((B_ * U_ + 255) / 256), dim3(256), 0, stream>>>(hf, out + (long)B_ * T_ * V_);
}
